// PropNet_35098472743097
// MI455X (gfx1250) — compile-verified, matrix operand order checked by reading, not run on hardware
//
#include <hip/hip_runtime.h>
#include <hip/hip_bf16.h>

typedef __attribute__((ext_vector_type(16))) _Float16 v16h;
typedef __attribute__((ext_vector_type(8)))  _Float16 v8h;
typedef __attribute__((ext_vector_type(8)))  float    v8f;

namespace {
constexpr int N_ = 2, C_ = 256, H_ = 96, W_ = 96;
constexpr int MD = 4, KD = 9, NC = 124;
constexpr int HW = H_ * W_;
constexpr int WT = W_ / 16;            // 6 pixel-tiles per image row
// Padded y geometry: rows +-4; cols +4 left, +12 right (tile1 reaches w0+27).
constexpr int PW = 112, PH = 104;
constexpr int PNP = PH * PW;           // padded pixels per image
constexpr size_t X16B  = (size_t)N_ * HW  * C_ * sizeof(_Float16); // 9.4 MB
constexpr size_t Y16PB = (size_t)N_ * PNP * C_ * sizeof(_Float16); // 11.9 MB
constexpr size_t N2B   = (size_t)N_ * HW  * sizeof(float);         // 73 KB
constexpr size_t P2B   = (size_t)N_ * PNP * sizeof(float);         // 93 KB
constexpr float PAD_BIG = 1e20f;

__device__ __forceinline__ float fast_rcp(float v) {
#if __has_builtin(__builtin_amdgcn_rcpf)
  return __builtin_amdgcn_rcpf(v);     // v_rcp_f32 (approx, sufficient here)
#else
  return 1.f / v;
#endif
}
}

// ---- Zero-fill padded fp16 buffer (16B stores).
__global__ __launch_bounds__(256)
void PropNet_zero_kernel(int4* __restrict__ p, int n16) {
  const int idx = blockIdx.x * blockDim.x + threadIdx.x;
  if (idx < n16) p[idx] = make_int4(0, 0, 0, 0);
}

// ---- NCHW f32 -> (optionally padded) NHWC fp16 via LDS tile transpose,
//      fused per-pixel squared-norm (fp32).
__global__ __launch_bounds__(256)
void PropNet_cvt_kernel(const float* __restrict__ src, _Float16* __restrict__ dst,
                        float* __restrict__ n2,
                        int nStridePix, int rowPix, int rowOff, int colOff) {
  __shared__ float tile[32][65];
  const int gp0 = blockIdx.x * 64;     // flat pixel base in [0, N*HW)
  const int tid = threadIdx.x;
  float sq = 0.f;                      // tid<64: running ||.||^2 of pixel gp0+tid
  for (int cpass = 0; cpass < 8; ++cpass) {
    const int c0 = cpass * 32;
    {   // read: lanes sweep pixels (contiguous w) -> coalesced
      const int pl = tid & 63;
      const int cl = tid >> 6;         // 0..3
      const int gp = gp0 + pl;
      const int n = gp / HW, pp = gp - n * HW;
      const float* s = src + (size_t)n * C_ * HW + pp;
#pragma unroll
      for (int i = 0; i < 8; ++i)
        tile[cl + i * 4][pl] = s[(size_t)(c0 + cl + i * 4) * HW];
    }
    __syncthreads();
    if (tid < 64) {                    // accumulate norm for owned pixel
      float s = 0.f;
#pragma unroll
      for (int c = 0; c < 32; ++c) { const float v = tile[c][tid]; s += v * v; }
      sq += s;
    }
    {   // write: lanes sweep channels -> contiguous fp16 in NHWC
      const int cl  = tid & 31;
      const int pl0 = tid >> 5;        // 0..7
#pragma unroll
      for (int j = 0; j < 8; ++j) {
        const int pl = pl0 + j * 8;
        const int gp = gp0 + pl;
        const int n = gp / HW, pp = gp - n * HW;
        const int h = pp / W_, w = pp - h * W_;
        const size_t dpix = (size_t)n * nStridePix
                          + (size_t)(h + rowOff) * rowPix + (w + colOff);
        dst[dpix * C_ + c0 + cl] = (_Float16)tile[cl][pl];
      }
    }
    __syncthreads();
  }
  if (tid < 64) n2[gp0 + tid] = sq;
}

// ---- Build padded y^2 (pad=1e20) and padded labels (pad=-1).
__global__ __launch_bounds__(256)
void PropNet_pad_kernel(const float* __restrict__ y2, const int* __restrict__ labels,
                        float* __restrict__ y2p, int* __restrict__ labp) {
  const int idx = blockIdx.x * blockDim.x + threadIdx.x;
  if (idx >= N_ * PNP) return;
  const int n = idx / PNP, pp = idx - n * PNP;
  const int hp = pp / PW, wp = pp - hp * PW;
  const int h = hp - MD, w = wp - MD;
  const bool in = (h >= 0) & (h < H_) & (w >= 0) & (w < W_);
  const size_t soff = (size_t)n * HW + (size_t)(in ? h : 0) * W_ + (in ? w : 0);
  y2p[idx]  = in ? y2[soff]     : PAD_BIG;
  labp[idx] = in ? labels[soff] : -1;
}

#define FRAG(p, kc)                                                         \
  __builtin_shufflevector((p)[(kc) * 4 + hi], (p)[(kc) * 4 + hi + 2],       \
                          0,1,2,3,4,5,6,7,8,9,10,11,12,13,14,15)

// ---- Main: 4 waves per (n, h, 16-pixel tile); each wave reduces a 64-channel
// K-slice on WMMA (padded fp16 NHWC, unconditional b128 loads). Partial 16x32
// D tiles summed in LDS; wave 0 does band extraction + sigmoid + class min.
__global__ __launch_bounds__(128)
void PropNet_main_kernel(const _Float16* __restrict__ x16,
                         const _Float16* __restrict__ y16p,
                         const float* __restrict__ x2buf,
                         const float* __restrict__ y2p,
                         const int* __restrict__ labp,
                         float* __restrict__ out) {
  const int tid  = threadIdx.x;
  const int u    = tid >> 5;           // wave id: K-slice [u*64, u*64+64)
  const int m    = tid & 15;           // pixel (A row) / B column within tile
  const int hi   = (tid >> 4) & 1;     // which K-half this lane carries

  int blk = blockIdx.x;
  const int wt = blk % WT; blk /= WT;
  const int h  = blk % H_;
  const int n  = blk / H_;
  const int w0 = wt * 16;

  __shared__ float mins[16][128];      // per-pixel class mins
  __shared__ float corrP[4][16][32];   // per-wave partial D tiles

  for (int i = tid; i < 16 * 128; i += 128) (&mins[0][0])[i] = 1.0f;

  // A fragments for this wave's two 32-channel chunks.
  const v8h* xa = (const v8h*)(x16 + ((size_t)n * HW + (size_t)h * W_ + (w0 + m)) * C_);
  const v16h a0 = FRAG(xa, 2 * u);
  const v16h a1 = FRAG(xa, 2 * u + 1);
  const float x2m = x2buf[(size_t)n * HW + (size_t)h * W_ + w0 + m];  // used by wave 0

  // Per-lane padded B columns: wp = w0 + m (+16); always in-bounds, pad = 0.
  const int wp0 = w0 + m;
  const int wp1 = wp0 + 16;
  const _Float16* ypn = y16p + (size_t)n * PNP * C_;
  const size_t dbase = (size_t)n * PNP;

  __syncthreads();

  for (int di = 0; di < KD; ++di) {
    const int hp = h + di;             // padded row index, in [0, PH)
    const v8h* yb0 = (const v8h*)(ypn + ((size_t)hp * PW + wp0) * C_);
    const v8h* yb1 = (const v8h*)(ypn + ((size_t)hp * PW + wp1) * C_);
    __builtin_prefetch(ypn + ((size_t)min(hp + 1, PH - 1) * PW + wp0) * C_, 0, 1);

    // 8 independent b128 loads, then 4 WMMAs (2 chunks x 2 tiles).
    const v16h b00 = FRAG(yb0, 2 * u);
    const v16h b10 = FRAG(yb1, 2 * u);
    const v16h b01 = FRAG(yb0, 2 * u + 1);
    const v16h b11 = FRAG(yb1, 2 * u + 1);
    v8f z = {};
    v8f s0 = __builtin_amdgcn_wmma_f32_16x16x32_f16(false, a0, false, b00,
                                                    (short)0, z, false, false);
    v8f s1 = __builtin_amdgcn_wmma_f32_16x16x32_f16(false, a0, false, b10,
                                                    (short)0, z, false, false);
    s0 = __builtin_amdgcn_wmma_f32_16x16x32_f16(false, a1, false, b01,
                                                (short)0, s0, false, false);
    s1 = __builtin_amdgcn_wmma_f32_16x16x32_f16(false, a1, false, b11,
                                                (short)0, s1, false, false);

    // Dump partial D tile: lane holds column m (+16), rows M = hi*8 + v.
#pragma unroll
    for (int v = 0; v < 8; ++v) {
      corrP[u][hi * 8 + v][m]      = s0[v];
      corrP[u][hi * 8 + v][16 + m] = s1[v];
    }
    __syncthreads();

    // Band phase (wave 0, lane m owns pixel m).
    if (tid < 16) {
      // Gather loop: 18 independent loads -> batched clauses, one wait.
      float y2a[KD]; int clsa[KD];
#pragma unroll
      for (int dj = 0; dj < KD; ++dj) {
        const size_t off = dbase + (size_t)hp * PW + (w0 + m + dj);
        y2a[dj]  = y2p[off];           // 1e20 in pad -> dv = 1.0
        clsa[dj] = labp[off];          // -1 in pad -> never matches
      }
      // Compute loop: (sigmoid(d)-0.5)*2 == (1-e)/(1+e), e = exp(-d) <= ~1.
#pragma unroll
      for (int dj = 0; dj < KD; ++dj) {
        const int b = m + dj;          // band column D[m, m+dj]
        const float cr = corrP[0][m][b] + corrP[1][m][b]
                       + corrP[2][m][b] + corrP[3][m][b];
        const float dd = x2m + y2a[dj] - 2.f * cr;
        const float e  = __expf(-dd);
        const float dv = (1.f - e) * fast_rcp(1.f + e);
        const int cls = clsa[dj];
        if ((unsigned)cls < (unsigned)NC)
          mins[m][cls] = fminf(mins[m][cls], dv);
      }
    }
    __syncthreads();
  }

  // out[n, cls, h, w0+mm]; consecutive tid sweep w -> 64B segments.
  float* ob = out + (size_t)n * NC * HW + (size_t)h * W_ + w0;
  for (int idx = tid; idx < NC * 16; idx += 128) {
    const int cls = idx >> 4, mm = idx & 15;
    ob[(size_t)cls * HW + mm] = mins[mm][cls];
  }
}

extern "C" void kernel_launch(void* const* d_in, const int* in_sizes, int n_in,
                              void* d_out, int out_size, void* d_ws, size_t ws_size,
                              hipStream_t stream) {
  (void)in_sizes; (void)n_in; (void)out_size; (void)ws_size;
  const float* x      = (const float*)d_in[0];
  const float* y      = (const float*)d_in[1];
  const int*   labels = (const int*)d_in[2];
  char* ws = (char*)d_ws;
  _Float16* x16  = (_Float16*)(ws);
  _Float16* y16p = (_Float16*)(ws + X16B);
  float*    x2b  = (float*)(ws + X16B + Y16PB);
  float*    y2b  = (float*)(ws + X16B + Y16PB + N2B);
  float*    y2p  = (float*)(ws + X16B + Y16PB + 2 * N2B);
  int*      labp = (int*)  (ws + X16B + Y16PB + 2 * N2B + P2B);
  float*    out  = (float*)d_out;

  const int npix  = N_ * HW;
  const int n16   = (int)(Y16PB / 16);
  PropNet_zero_kernel<<<(n16 + 255) / 256, 256, 0, stream>>>((int4*)y16p, n16);
  PropNet_cvt_kernel<<<npix / 64, 256, 0, stream>>>(x, x16, x2b, HW, W_, 0, 0);
  PropNet_cvt_kernel<<<npix / 64, 256, 0, stream>>>(y, y16p, y2b, PNP, PW, MD, MD);
  PropNet_pad_kernel<<<(N_ * PNP + 255) / 256, 256, 0, stream>>>(y2b, labels, y2p, labp);
  PropNet_main_kernel<<<N_ * H_ * WT, 128, 0, stream>>>(x16, y16p, x2b, y2p, labp, out);
}
